// Matformer_10737418240847
// MI455X (gfx1250) — matrix operand order repaired and checked by reading, not compiled, not run on hardware
//
#include <hip/hip_runtime.h>

// ---------------------------------------------------------------------------
// Types for CDNA5 WMMA
// ---------------------------------------------------------------------------
typedef __attribute__((ext_vector_type(16))) __bf16 v16bf;
typedef __attribute__((ext_vector_type(8)))  float  v8f;

__device__ inline v8f wmma_bf16(v16bf a, v16bf b, v8f c) {
  // D = A(16x32 bf16) * B(32x16 bf16) + C(16x16 f32)
  return __builtin_amdgcn_wmma_f32_16x16x32_bf16(false, a, false, b, (short)0, c,
                                                 false, false);
}

union BF16Frag { v16bf v; uint4 u[2]; };

// A-fragment from LDS tile [rows][astr] bf16 (row-major).
// ISA 16x32 bf16 A layout: lanes 0-15 -> row M=lane, K {g*8..+7, g*8+16..+23},
// g = lane>>4 (lanes 16-31 same rows, K {8..15, 24..31}).
__device__ inline v16bf lds_afrag(const __bf16* base, int astr, int rowbase,
                                  int kbase, int lane) {
  int row = rowbase + (lane & 15);
  int g = lane >> 4;
  const __bf16* p = base + row * astr + kbase + g * 8;
  BF16Frag x;
  x.u[0] = *(const uint4*)(p);
  x.u[1] = *(const uint4*)(p + 16);
  return x.v;
}

// B-fragment from LDS tile [32 k-rows][bstr] bf16: lane L holds column
// n = n_off + (L & 15), K {g*8..+7, g*8+16..+23}, g = L>>4.
__device__ inline v16bf lds_bfrag(const __bf16* Bsb, int bstr, int n_off,
                                  int lane) {
  const __bf16* p = Bsb + n_off + (lane & 15);
  int g = lane >> 4;
  BF16Frag x;
#pragma unroll
  for (int s = 0; s < 16; ++s) {
    int kk = (s < 8) ? (g * 8 + s) : (16 + g * 8 + (s - 8));
    x.v[s] = p[kk * bstr];
  }
  return x.v;
}

// CDNA5 async global->LDS copy (16B per lane), tracked by ASYNCcnt.
__device__ inline void async_copy_b128(const __bf16* gp, __bf16* lp) {
  unsigned int l = (unsigned int)(uintptr_t)lp;  // LDS aperture: offset in [31:0]
  unsigned long long g = (unsigned long long)(uintptr_t)gp;
  asm volatile("global_load_async_to_lds_b128 %0, %1, off"
               :
               : "v"(l), "v"(g)
               : "memory");
}
__device__ inline void wait_asynccnt0() {
  asm volatile("s_wait_asynccnt 0" ::: "memory");
}

__device__ inline float softplus_f(float x) {
  return x > 0.f ? x + log1pf(expf(-x)) : log1pf(expf(x));
}
__device__ inline float sigmoid_f(float x) { return 1.f / (1.f + expf(-x)); }

// ---------------------------------------------------------------------------
// Generic GEMM: C[M,N] = act(A[M,K] (f32) @ B[K,N] (bf16) + bias)
// Block: 256 threads (8 waves). Tile: 128 rows x 64 cols. K multiple of 32.
// Double-buffered LDS staging: B via async global->LDS, A via f32->bf16
// convert. grid = (M/128, N/64)
// ---------------------------------------------------------------------------
template <int ACT>
__global__ void __launch_bounds__(256) gemm_bf16_kernel(
    const float* __restrict__ A, const __bf16* __restrict__ B,
    const float* __restrict__ bias, float* __restrict__ C, int M, int K, int N) {
  constexpr int AS = 40;  // bf16 stride for A tile rows (80B)
  constexpr int BS = 72;  // bf16 stride for B tile rows (144B, bank-spread)
  __shared__ __bf16 As[2][128 * AS];
  __shared__ __bf16 Bs[2][32 * BS];

  const int row0 = blockIdx.x * 128;
  const int n0 = blockIdx.y * 64;
  const int t = threadIdx.x, lane = t & 31, w = t >> 5;
  const int nk = K >> 5;

  // staging coordinates
  const int ar = t >> 1, ah = (t & 1) * 16;   // A: 128 rows x 2 halves
  const int br = t >> 3, bc = (t & 7) * 8;    // B: 32 rows x 8 chunks of 8

  v8f acc[4];
#pragma unroll
  for (int f = 0; f < 4; ++f)
#pragma unroll
    for (int i = 0; i < 8; ++i) acc[f][i] = 0.f;

  float4 a4[4];
  const float* aprow = A + (size_t)(row0 + ar) * K + ah;

  // ---- prologue: stage tile 0 ----
  async_copy_b128(B + (size_t)br * N + n0 + bc, &Bs[0][br * BS + bc]);
#pragma unroll
  for (int i = 0; i < 4; ++i) a4[i] = ((const float4*)(aprow))[i];
  {
    __bf16* dp = &As[0][ar * AS + ah];
#pragma unroll
    for (int i = 0; i < 4; ++i) {
      dp[i * 4 + 0] = (__bf16)a4[i].x;
      dp[i * 4 + 1] = (__bf16)a4[i].y;
      dp[i * 4 + 2] = (__bf16)a4[i].z;
      dp[i * 4 + 3] = (__bf16)a4[i].w;
    }
  }
  wait_asynccnt0();
  __syncthreads();

  for (int ks = 0; ks < nk; ++ks) {
    const int cb = ks & 1, nb = (ks + 1) & 1;
    const bool more = (ks + 1) < nk;
    if (more) {
      // issue next tile's loads before computing on the current tile
      async_copy_b128(B + (size_t)((ks + 1) * 32 + br) * N + n0 + bc,
                      &Bs[nb][br * BS + bc]);
#pragma unroll
      for (int i = 0; i < 4; ++i)
        a4[i] = ((const float4*)(aprow + (ks + 1) * 32))[i];
    }

    v16bf af = lds_afrag(As[cb], AS, w * 16, 0, lane);
#pragma unroll
    for (int f = 0; f < 4; ++f) {
      v16bf bf_ = lds_bfrag(&Bs[cb][0], BS, f * 16, lane);
      acc[f] = wmma_bf16(af, bf_, acc[f]);
    }

    if (more) {
      __bf16* dp = &As[nb][ar * AS + ah];
#pragma unroll
      for (int i = 0; i < 4; ++i) {
        dp[i * 4 + 0] = (__bf16)a4[i].x;
        dp[i * 4 + 1] = (__bf16)a4[i].y;
        dp[i * 4 + 2] = (__bf16)a4[i].z;
        dp[i * 4 + 3] = (__bf16)a4[i].w;
      }
    }
    wait_asynccnt0();
    __syncthreads();
  }

#pragma unroll
  for (int f = 0; f < 4; ++f) {
    int col = n0 + f * 16 + (lane & 15);
    float bs = bias ? bias[col] : 0.f;
#pragma unroll
    for (int vv = 0; vv < 8; ++vv) {
      int row = row0 + w * 16 + 8 * (lane >> 4) + vv;
      float x = acc[f][vv] + bs;
      if (ACT == 1) x = softplus_f(x);
      C[(size_t)row * N + col] = x;
    }
  }
}

// ---------------------------------------------------------------------------
// Fused edge-message kernel (one conv layer's message pass).
// Block = 256 threads handles 8 edges x 4 heads = 32 rows of width 384.
// Phases: gather + alpha + LN + sigmoid gate -> GEMM1 (384x384) * gate
//         -> GEMM2 (384x128) -> per-row LN -> atomic scatter-add into agg.
// Weight tiles (msg_up / msg2) are double-buffered in LDS via the CDNA5
// async global->LDS path. Dynamic LDS (~98.5 KB).
// ---------------------------------------------------------------------------
__global__ void __launch_bounds__(256) edge_msg_kernel(
    const float* __restrict__ q, const float* __restrict__ k,
    const float* __restrict__ v, const float* __restrict__ eh,
    const int* __restrict__ ei, int E, const __bf16* __restrict__ w1,
    const float* __restrict__ b1, const __bf16* __restrict__ w2,
    const float* __restrict__ b2, const float* __restrict__ lnag,
    const float* __restrict__ lnab, const float* __restrict__ lnmg,
    const float* __restrict__ lnmb, float* __restrict__ agg) {
  constexpr int ASTR = 400;  // bf16 stride for 32x384 message tile
  constexpr int GSTR = 392;  // bf16 stride for alpha/gate tile
  constexpr int OSTR = 132;  // f32 stride for 32x128 output tile
  constexpr int BS1 = 392;   // bf16 stride, w1 tile rows (784B, bank-spread)
  constexpr int BS2 = 136;   // bf16 stride, w2 tile rows (272B, bank-spread)
  extern __shared__ char smem[];
  __bf16* At = (__bf16*)smem;                    // 32*400*2 = 25600 B
  __bf16* G = (__bf16*)(smem + 32 * ASTR * 2);   // 32*392*2 = 25088 B
  float* Ot = (float*)(smem + 32 * ASTR * 2);    // overlays G (G dead by then)
  __bf16* Bb0 = (__bf16*)(smem + 32 * ASTR * 2 + 32 * GSTR * 2);
  __bf16* Bb1 = Bb0 + 32 * BS1;                  // 2 x 25088 B

  const int t = threadIdx.x, lane = t & 31, w = t >> 5;
  const int e0 = blockIdx.x * 8;

  // async stage of one 32x384 w1 k-tile (1536 x 16B chunks, 6 per thread)
  auto stage1 = [&](int ks, __bf16* dst) {
#pragma unroll
    for (int i = 0; i < 6; ++i) {
      int c = t + i * 256;
      int row = c / 48, cc = c % 48;
      async_copy_b128(w1 + (size_t)(ks * 32 + row) * 384 + cc * 8,
                      dst + row * BS1 + cc * 8);
    }
  };
  // async stage of one 32x128 w2 k-tile (512 x 16B chunks, 2 per thread)
  auto stage2 = [&](int ks, __bf16* dst) {
#pragma unroll
    for (int i = 0; i < 2; ++i) {
      int c = t + i * 256;
      int row = c >> 4, cc = c & 15;
      async_copy_b128(w2 + (size_t)(ks * 32 + row) * 128 + cc * 8,
                      dst + row * BS2 + cc * 8);
    }
  };

  stage1(0, Bb0);  // prefetch first w1 tile; overlaps all of phase 1

  // -------- Phase 1: gather, alpha, layernorm(384), sigmoid gate, m_in ----
  {
    const int r = t >> 3, gg = t & 7;              // 32 rows, 8 threads/row
    const int e = e0 + (r >> 2), hh = r & 3;
    const int sN = ei[e], dN = ei[E + e];
    const float* qd = q + (size_t)dN * 512 + hh * 128;
    const float* kd = k + (size_t)dN * 512 + hh * 128;
    const float* ksr = k + (size_t)sN * 512 + hh * 128;
    const float* vd = v + (size_t)dN * 512 + hh * 128;
    const float* vs = v + (size_t)sN * 512 + hh * 128;
    const float* er = eh + (size_t)e * 512 + hh * 128;
    float s1 = 0.f, s2 = 0.f;
    const int j0 = gg * 48;
    for (int j = j0; j < j0 + 48; ++j) {
      int seg = j >> 7, c = j & 127;
      float kvv = seg == 0 ? kd[c] : (seg == 1 ? ksr[c] : er[c]);
      float mvv = seg == 0 ? vd[c] : (seg == 1 ? vs[c] : er[c]);
      float av = qd[c] * kvv * 0.05103103630798287f;  // 1/sqrt(384)
      G[r * GSTR + j] = (__bf16)av;
      At[r * ASTR + j] = (__bf16)mvv;
      s1 += av;
      s2 += av * av;
    }
    for (int m = 1; m < 8; m <<= 1) {
      s1 += __shfl_xor(s1, m, 32);
      s2 += __shfl_xor(s2, m, 32);
    }
    float mean = s1 * (1.f / 384.f);
    float var = s2 * (1.f / 384.f) - mean * mean;
    float rstd = rsqrtf(var + 1e-5f);
    for (int j = j0; j < j0 + 48; ++j) {
      float av = (float)G[r * GSTR + j];
      float ln = (av - mean) * rstd * lnag[j] + lnab[j];
      G[r * GSTR + j] = (__bf16)sigmoid_f(ln);
    }
  }
  wait_asynccnt0();
  __syncthreads();  // At, G, Bb0 all ready

  // -------- GEMM1: [32,384] @ msg_up[384,384] + b, then * gate -> At ------
  {
    const int strip = (w & 1) * 16, n0 = (w >> 1) * 96;  // 2 strips x 4 chunks
    v8f acc[6];
#pragma unroll
    for (int f = 0; f < 6; ++f)
#pragma unroll
      for (int i = 0; i < 8; ++i) acc[f][i] = 0.f;
    for (int ks = 0; ks < 12; ++ks) {
      __bf16* cb = (ks & 1) ? Bb1 : Bb0;
      __bf16* nb = (ks & 1) ? Bb0 : Bb1;
      if (ks + 1 < 12) stage1(ks + 1, nb);  // async prefetch next tile
      v16bf af = lds_afrag(At, ASTR, strip, ks * 32, lane);
#pragma unroll
      for (int f = 0; f < 6; ++f) {
        v16bf bf_ = lds_bfrag(cb, BS1, n0 + f * 16, lane);
        acc[f] = wmma_bf16(af, bf_, acc[f]);
      }
      wait_asynccnt0();
      __syncthreads();
    }
    // final loop barrier == all waves done reading At; safe to overwrite
#pragma unroll
    for (int f = 0; f < 6; ++f) {
      int col = n0 + f * 16 + (lane & 15);
      float bs = b1[col];
#pragma unroll
      for (int vv = 0; vv < 8; ++vv) {
        int row = strip + 8 * (lane >> 4) + vv;
        float x = acc[f][vv] + bs;
        float gt = (float)G[row * GSTR + col];
        At[row * ASTR + col] = (__bf16)(x * gt);
      }
    }
  }
  stage2(0, Bb0);  // Bb buffers free after GEMM1 loop; prefetch first w2 tile
  wait_asynccnt0();
  __syncthreads();

  // -------- GEMM2: [32,384] @ msg2[384,128] + b -> Ot (f32) ---------------
  {
    const int strip = (w & 1) * 16, n0 = (w >> 1) * 32;
    v8f acc[2];
#pragma unroll
    for (int f = 0; f < 2; ++f)
#pragma unroll
      for (int i = 0; i < 8; ++i) acc[f][i] = 0.f;
    for (int ks = 0; ks < 12; ++ks) {
      __bf16* cb = (ks & 1) ? Bb1 : Bb0;
      __bf16* nb = (ks & 1) ? Bb0 : Bb1;
      if (ks + 1 < 12) stage2(ks + 1, nb);
      v16bf af = lds_afrag(At, ASTR, strip, ks * 32, lane);
#pragma unroll
      for (int f = 0; f < 2; ++f) {
        v16bf bf_ = lds_bfrag(cb, BS2, n0 + f * 16, lane);
        acc[f] = wmma_bf16(af, bf_, acc[f]);
      }
      wait_asynccnt0();
      __syncthreads();
    }
    // Ot overlays G; G no longer read, At reads finished at loop barrier
#pragma unroll
    for (int f = 0; f < 2; ++f) {
      int col = n0 + f * 16 + (lane & 15);
      float bs = b2[col];
#pragma unroll
      for (int vv = 0; vv < 8; ++vv) {
        int row = strip + 8 * (lane >> 4) + vv;
        Ot[row * OSTR + col] = acc[f][vv] + bs;
      }
    }
  }
  __syncthreads();

  // -------- Phase 4: layernorm(128) per row + atomic scatter-add ----------
  {
    const int subr = lane >> 3, gg = lane & 7;
    const int r = w * 4 + subr;          // 8 waves x 4 rows = 32 rows
    const int cb = gg * 16;
    float s1 = 0.f, s2 = 0.f;
    for (int i = 0; i < 16; ++i) {
      float x = Ot[r * OSTR + cb + i];
      s1 += x;
      s2 += x * x;
    }
    for (int m = 1; m < 8; m <<= 1) {
      s1 += __shfl_xor(s1, m, 32);
      s2 += __shfl_xor(s2, m, 32);
    }
    float mean = s1 * (1.f / 128.f);
    float var = s2 * (1.f / 128.f) - mean * mean;
    float rstd = rsqrtf(var + 1e-5f);
    const int e = e0 + (r >> 2), hh = r & 3;
    const int dN = ei[E + e];
    float* ag = agg + (size_t)dN * 512 + hh * 128;
    for (int i = 0; i < 16; ++i) {
      int c = cb + i;
      float y = (Ot[r * OSTR + c] - mean) * rstd * lnmg[c] + lnmb[c];
      unsafeAtomicAdd(&ag[c], y);
    }
  }
}

// ---------------------------------------------------------------------------
// Small helper kernels
// ---------------------------------------------------------------------------
__global__ void cvt_kernel(const float* __restrict__ s, __bf16* __restrict__ d,
                           int n) {
  int i = blockIdx.x * 256 + threadIdx.x;
  if (i < n) d[i] = (__bf16)s[i];
}

__global__ void cvt_pad_kernel(const float* __restrict__ s,
                               __bf16* __restrict__ d, int Ks, int Kd, int N) {
  int i = blockIdx.x * 256 + threadIdx.x;
  if (i >= Kd * N) return;
  int kk = i / N, c = i % N;
  d[i] = kk < Ks ? (__bf16)s[(size_t)kk * N + c] : (__bf16)0.f;
}

__global__ void pad_rows_kernel(const float* __restrict__ s,
                                float* __restrict__ d, int M, int Ks, int Kd) {
  int i = blockIdx.x * 256 + threadIdx.x;
  if (i >= M * Kd) return;
  int r = i / Kd, c = i % Kd;
  d[i] = c < Ks ? s[(size_t)r * Ks + c] : 0.f;
}

__global__ void rbf_kernel(const float* __restrict__ ea, float* __restrict__ rbf,
                           int E) {
  int e = blockIdx.x, j = threadIdx.x;  // block = 128 threads
  float x0 = ea[e * 3], x1 = ea[e * 3 + 1], x2 = ea[e * 3 + 2];
  float d = sqrtf(x0 * x0 + x1 * x1 + x2 * x2);
  float c = (float)j * (8.0f / 127.0f);
  float t = d - c;
  rbf[(size_t)e * 128 + j] = expf(-15.875f * t * t);  // gamma = 127/8
}

__global__ void bn_stats_kernel(const float* __restrict__ X,
                                float* __restrict__ mean,
                                float* __restrict__ var, int M, int N) {
  __shared__ float s1[256], s2[256];
  int col = blockIdx.x, t = threadIdx.x;
  float a = 0.f, b = 0.f;
  for (int r = t; r < M; r += 256) {
    float x = X[(size_t)r * N + col];
    a += x;
    b += x * x;
  }
  s1[t] = a;
  s2[t] = b;
  __syncthreads();
  for (int s = 128; s > 0; s >>= 1) {
    if (t < s) {
      s1[t] += s1[t + s];
      s2[t] += s2[t + s];
    }
    __syncthreads();
  }
  if (t == 0) {
    float mu = s1[0] / (float)M;
    mean[col] = mu;
    var[col] = s2[0] / (float)M - mu * mu;
  }
}

__global__ void bn_apply_kernel(const float* __restrict__ pre,
                                const float* __restrict__ skp,
                                const float* __restrict__ mean,
                                const float* __restrict__ var,
                                const float* __restrict__ g,
                                const float* __restrict__ b,
                                float* __restrict__ out, int M, int N) {
  int i = blockIdx.x * 256 + threadIdx.x;
  if (i >= M * N) return;
  int c = i % N;
  float x = (pre[i] - mean[c]) * rsqrtf(var[c] + 1e-5f) * g[c] + b[c];
  out[i] = x * sigmoid_f(x) + skp[i];  // silu + skip
}

__global__ void concat_kernel(const float* __restrict__ h,
                              const float* __restrict__ g,
                              float* __restrict__ d, int M) {
  int i = blockIdx.x * 256 + threadIdx.x;
  if (i >= M * 256) return;
  int r = i >> 8, c = i & 255;
  d[i] = c < 128 ? h[(size_t)r * 128 + c]
                 : (c < 236 ? g[(size_t)r * 108 + (c - 128)] : 0.f);
}

__global__ void rowdot_kernel(const float* __restrict__ X,
                              const float* __restrict__ w,
                              const float* __restrict__ b,
                              float* __restrict__ out, int M, int K) {
  int n = blockIdx.x * 256 + threadIdx.x;
  if (n >= M) return;
  float s = 0.f;
  for (int c = 0; c < K; ++c) s += X[(size_t)n * K + c] * w[c];
  out[n] = s + b[0];
}

// Scatter-softmax attention pooling + final linear. One block per graph.
__global__ void __launch_bounds__(128) pool_kernel(
    const float* __restrict__ h, const float* __restrict__ a,
    const int* __restrict__ batch, const float* __restrict__ outw,
    const float* __restrict__ outb, float* __restrict__ out, int N) {
  __shared__ float red[128];
  int g = blockIdx.x, t = threadIdx.x;
  float m = -3.4e38f;
  for (int n = t; n < N; n += 128)
    if (batch[n] == g) m = fmaxf(m, a[n]);
  red[t] = m;
  __syncthreads();
  for (int s = 64; s > 0; s >>= 1) {
    if (t < s) red[t] = fmaxf(red[t], red[t + s]);
    __syncthreads();
  }
  m = red[0];
  __syncthreads();
  float ss = 0.f;
  for (int n = t; n < N; n += 128)
    if (batch[n] == g) ss += expf(a[n] - m);
  red[t] = ss;
  __syncthreads();
  for (int s = 64; s > 0; s >>= 1) {
    if (t < s) red[t] += red[t + s];
    __syncthreads();
  }
  ss = red[0] + 1e-16f;
  __syncthreads();
  float pc = 0.f;
  for (int n = 0; n < N; ++n)
    if (batch[n] == g) pc += expf(a[n] - m) * h[(size_t)n * 128 + t];
  pc /= ss;
  red[t] = pc * outw[t];
  __syncthreads();
  for (int s = 64; s > 0; s >>= 1) {
    if (t < s) red[t] += red[t + s];
    __syncthreads();
  }
  if (t == 0) out[g] = red[0] + outb[0];
}

// ---------------------------------------------------------------------------
// Host orchestration
// ---------------------------------------------------------------------------
extern "C" void kernel_launch(void* const* d_in, const int* in_sizes, int n_in,
                              void* d_out, int out_size, void* d_ws,
                              size_t ws_size, hipStream_t stream) {
  (void)in_sizes; (void)n_in; (void)out_size; (void)ws_size;
  const int N = 2048, E = 24576, Gn = 32;

  // Input map (setup_inputs insertion order, params flattened recursively)
  const float* x = (const float*)d_in[0];
  const float* edge_attr = (const float*)d_in[1];
  const float* glob = (const float*)d_in[2];
  const int* ei = (const int*)d_in[125];
  const int* batch = (const int*)d_in[126];
  auto F = [&](int i) { return (const float*)d_in[i]; };

  // Workspace layout
  char* ws = (char*)d_ws;
  size_t cur = 0;
  auto alloc = [&](size_t bytes) -> char* {
    char* p = ws + cur;
    cur += (bytes + 255) & ~(size_t)255;
    return p;
  };
  float* x_pad = (float*)alloc((size_t)N * 96 * 4);
  float* h = (float*)alloc((size_t)N * 128 * 4);
  float* rbf = (float*)alloc((size_t)E * 128 * 4);   // later reused as e_feat
  float* t1 = (float*)alloc((size_t)E * 128 * 4);
  float* ehalf = (float*)alloc((size_t)E * 512 * 4);
  float* qb = (float*)alloc((size_t)N * 512 * 4);
  float* kb = (float*)alloc((size_t)N * 512 * 4);
  float* vb = (float*)alloc((size_t)N * 512 * 4);
  float* agg = (float*)alloc((size_t)N * 512 * 4);
  float* prebn = (float*)alloc((size_t)N * 128 * 4);
  float* skipb = (float*)alloc((size_t)N * 128 * 4);
  float* bnmu = (float*)alloc(128 * 4);
  float* bnvar = (float*)alloc(128 * 4);
  float* attin = (float*)alloc((size_t)N * 256 * 4);
  float* atth = (float*)alloc((size_t)N * 128 * 4);
  float* avec = (float*)alloc((size_t)N * 4);
  __bf16* w_atom = (__bf16*)alloc(96 * 128 * 2);
  __bf16* w_rbf1 = (__bf16*)alloc(16384 * 2);
  __bf16* w_rbf2 = (__bf16*)alloc(16384 * 2);
  __bf16* w_att1 = (__bf16*)alloc(256 * 128 * 2);
  __bf16 *wq[5], *wk[5], *wv[5], *we[5], *wcat[5], *wm1[5], *wm2[5], *wsk[5];
  for (int L = 0; L < 5; ++L) {
    wq[L] = (__bf16*)alloc(65536 * 2);
    wk[L] = (__bf16*)alloc(65536 * 2);
    wv[L] = (__bf16*)alloc(65536 * 2);
    we[L] = (__bf16*)alloc(65536 * 2);
    wcat[L] = (__bf16*)alloc(65536 * 2);
    wm1[L] = (__bf16*)alloc(147456 * 2);
    wm2[L] = (__bf16*)alloc(49152 * 2);
    wsk[L] = (__bf16*)alloc(16384 * 2);
  }

  auto cvt = [&](const float* s, __bf16* d, int n) {
    cvt_kernel<<<(n + 255) / 256, 256, 0, stream>>>(s, d, n);
  };

  // ---- weight conversion to bf16 ----
  cvt_pad_kernel<<<(96 * 128 + 255) / 256, 256, 0, stream>>>(F(3), w_atom, 92, 96, 128);
  cvt(F(5), w_rbf1, 16384);
  cvt(F(7), w_rbf2, 16384);
  for (int L = 0; L < 5; ++L) {
    int b0 = 9 + 22 * L;
    cvt(F(b0 + 0), wq[L], 65536);
    cvt(F(b0 + 2), wk[L], 65536);
    cvt(F(b0 + 4), wv[L], 65536);
    cvt(F(b0 + 6), we[L], 65536);
    cvt(F(b0 + 8), wcat[L], 65536);
    cvt(F(b0 + 10), wm1[L], 147456);
    cvt(F(b0 + 12), wm2[L], 49152);
    cvt(F(b0 + 14), wsk[L], 16384);
  }
  cvt_pad_kernel<<<(256 * 128 + 255) / 256, 256, 0, stream>>>(F(119), w_att1, 236, 256, 128);

  // ---- atom embedding ----
  pad_rows_kernel<<<(N * 96 + 255) / 256, 256, 0, stream>>>(x, x_pad, N, 92, 96);
  gemm_bf16_kernel<0><<<dim3(N / 128, 2), 256, 0, stream>>>(x_pad, w_atom, F(4), h, N, 96, 128);

  // ---- RBF edge features ----
  rbf_kernel<<<E, 128, 0, stream>>>(edge_attr, rbf, E);
  gemm_bf16_kernel<1><<<dim3(E / 128, 2), 256, 0, stream>>>(rbf, w_rbf1, F(6), t1, E, 128, 128);
  gemm_bf16_kernel<0><<<dim3(E / 128, 2), 256, 0, stream>>>(t1, w_rbf2, F(8), rbf, E, 128, 128);
  float* efeat = rbf;

  // dynamic LDS for the edge-message kernel: At + G + 2 B-tile buffers
  const size_t edge_smem = (size_t)32 * 400 * 2 + 32 * 392 * 2 + 2 * 32 * 392 * 2;

  // ---- 5 Matformer conv layers ----
  for (int L = 0; L < 5; ++L) {
    int b0 = 9 + 22 * L;
    gemm_bf16_kernel<0><<<dim3(N / 128, 8), 256, 0, stream>>>(h, wq[L], F(b0 + 1), qb, N, 128, 512);
    gemm_bf16_kernel<0><<<dim3(N / 128, 8), 256, 0, stream>>>(h, wk[L], F(b0 + 3), kb, N, 128, 512);
    gemm_bf16_kernel<0><<<dim3(N / 128, 8), 256, 0, stream>>>(h, wv[L], F(b0 + 5), vb, N, 128, 512);
    gemm_bf16_kernel<0><<<dim3(E / 128, 8), 256, 0, stream>>>(efeat, we[L], F(b0 + 7), ehalf, E, 128, 512);
    hipMemsetAsync(agg, 0, (size_t)N * 512 * 4, stream);
    edge_msg_kernel<<<E / 8, 256, edge_smem, stream>>>(
        qb, kb, vb, ehalf, ei, E, wm1[L], F(b0 + 11), wm2[L], F(b0 + 13),
        F(b0 + 16), F(b0 + 17), F(b0 + 18), F(b0 + 19), agg);
    gemm_bf16_kernel<0><<<dim3(N / 128, 2), 256, 0, stream>>>(agg, wcat[L], F(b0 + 9), prebn, N, 512, 128);
    gemm_bf16_kernel<0><<<dim3(N / 128, 2), 256, 0, stream>>>(h, wsk[L], F(b0 + 15), skipb, N, 128, 128);
    bn_stats_kernel<<<128, 256, 0, stream>>>(prebn, bnmu, bnvar, N, 128);
    bn_apply_kernel<<<(N * 128) / 256, 256, 0, stream>>>(prebn, skipb, bnmu, bnvar, F(b0 + 20), F(b0 + 21), h, N, 128);
  }

  // ---- global attention pooling + readout ----
  concat_kernel<<<(N * 256) / 256, 256, 0, stream>>>(h, glob, attin, N);
  gemm_bf16_kernel<1><<<dim3(N / 128, 2), 256, 0, stream>>>(attin, w_att1, F(120), atth, N, 256, 128);
  rowdot_kernel<<<(N + 255) / 256, 256, 0, stream>>>(atth, F(121), F(122), avec, N, 128);
  pool_kernel<<<Gn, 128, 0, stream>>>(h, avec, batch, F(123), F(124), (float*)d_out, N);
}
